// Encoder_76948634075520
// MI455X (gfx1250) — matrix operand order repaired and checked by reading, not compiled, not run on hardware
//
#include <hip/hip_runtime.h>

// Problem constants (fixed by the reference)
static constexpr int BB = 128;   // batch
static constexpr int TT = 512;   // time steps
static constexpr int NN = 12;    // features (= H)
static constexpr int GG = 48;    // 4*H gates
static constexpr int QQ = 24;    // query width = 2*H

typedef __attribute__((ext_vector_type(16))) __bf16 v16bf;
typedef __attribute__((ext_vector_type(8)))  float  v8f;

__device__ __forceinline__ float fast_rcp(float x) {
    return __builtin_amdgcn_rcpf(x);
}
__device__ __forceinline__ float sigmoid_f(float x) {
    return fast_rcp(1.0f + __expf(-x));
}
__device__ __forceinline__ float tanh_f(float x) {
    x = fminf(8.0f, fmaxf(-8.0f, x));
    float e = __expf(2.0f * x);
    return (e - 1.0f) * fast_rcp(e + 1.0f);
}

// ---------------------------------------------------------------------------
// Kernel 1: two zero-state LSTM steps per (b,t) -> query = [h2, c2]  (B,T,24)
// ---------------------------------------------------------------------------
__global__ void lstm_query_kernel(const float* __restrict__ data,
                                  const float* __restrict__ W_ih1,
                                  const float* __restrict__ b_ih1,
                                  const float* __restrict__ b_hh1,
                                  const float* __restrict__ W_ih2,
                                  const float* __restrict__ b_ih2,
                                  const float* __restrict__ b_hh2,
                                  float* __restrict__ query) {
    __shared__ float sW1[GG * NN];
    __shared__ float sW2[GG * NN];
    __shared__ float sb1[GG];
    __shared__ float sb2[GG];
    const int tid = threadIdx.x;
    for (int i = tid; i < GG * NN * 2 + 2 * GG; i += blockDim.x) {
        if (i < GG * NN) {
            sW1[i] = W_ih1[i];
        } else if (i < 2 * GG * NN) {
            sW2[i - GG * NN] = W_ih2[i - GG * NN];
        } else if (i < 2 * GG * NN + GG) {
            int g = i - 2 * GG * NN;
            sb1[g] = b_ih1[g] + b_hh1[g];   // h0 == 0, so W_hh never matters
        } else {
            int g = i - 2 * GG * NN - GG;
            sb2[g] = b_ih2[g] + b_hh2[g];
        }
    }
    __syncthreads();

    const int bt = blockIdx.x * blockDim.x + tid;
    if (bt >= BB * TT) return;

    float x[NN];
#pragma unroll
    for (int j = 0; j < NN; ++j) x[j] = data[bt * NN + j];

    // layer 1
    float g1[GG];
#pragma unroll
    for (int g = 0; g < GG; ++g) {
        float acc = sb1[g];
#pragma unroll
        for (int j = 0; j < NN; ++j) acc = __fmaf_rn(sW1[g * NN + j], x[j], acc);
        g1[g] = acc;
    }
    float h1[NN];
#pragma unroll
    for (int k = 0; k < NN; ++k) {
        float c = sigmoid_f(g1[k]) * tanh_f(g1[24 + k]);      // i*tanh(g); f*c0 = 0
        h1[k] = sigmoid_f(g1[36 + k]) * tanh_f(c);            // o*tanh(c)
    }

    // layer 2
    float g2[GG];
#pragma unroll
    for (int g = 0; g < GG; ++g) {
        float acc = sb2[g];
#pragma unroll
        for (int j = 0; j < NN; ++j) acc = __fmaf_rn(sW2[g * NN + j], h1[j], acc);
        g2[g] = acc;
    }
#pragma unroll
    for (int k = 0; k < NN; ++k) {
        float c2 = sigmoid_f(g2[k]) * tanh_f(g2[24 + k]);
        float h2 = sigmoid_f(g2[36 + k]) * tanh_f(c2);
        query[bt * QQ + k]      = h2;
        query[bt * QQ + NN + k] = c2;
    }
}

// ---------------------------------------------------------------------------
// Kernel 2: w1x[b,m,u] = sum_t data[b,t,m] * W1[u,t] + b1[u]   (bf16 WMMA)
//   per-batch GEMM: M=16 (12 valid, zero-padded in LDS), K=512, N=512.
//   grid = B*8 blocks of 128 threads; each wave owns one 16-wide u tile.
//   data[b] staged TRANSPOSED in LDS so each lane's A elements are contiguous
//   (vectorizable ds loads, no per-lane conditionals -> no exec branching).
// ---------------------------------------------------------------------------
__global__ void w1x_kernel(const float* __restrict__ data,
                           const float* __restrict__ W1,
                           const float* __restrict__ b1,
                           float* __restrict__ w1x) {
    __shared__ float s_dataT[16 * 520];            // [m][t], rows 12..15 stay zero
    const int tid = threadIdx.x;
    const int b = blockIdx.x >> 3;                 // 8 blocks per batch
    for (int i = tid; i < 16 * 520; i += blockDim.x) s_dataT[i] = 0.0f;
    __syncthreads();
    for (int i = tid; i < TT * NN; i += blockDim.x) {
        int t = i / NN, mm = i - t * NN;
        s_dataT[mm * 520 + t] = data[b * TT * NN + i];
    }
    __syncthreads();

    const int lane   = tid & 31;
    const int warp   = tid >> 5;
    const int u_tile = ((blockIdx.x & 7) << 2) | warp;   // 0..31
    const int u_col  = u_tile * 16 + (lane & 15);
    const bool hi    = lane >= 16;
    const int m      = lane & 15;                  // A-matrix row (feature index)
    const int akb0   = hi ? 16 : 0;                // A frag K sub-ranges: contiguous k = 16h + e
    const int akb1   = hi ? 24 : 8;
    const int bkb    = hi ? 16 : 0;                // B frag K range: contiguous k = 16h + e

    const float* __restrict__ arow = &s_dataT[m * 520];

    v8f acc = {};
    for (int k0 = 0; k0 < TT; k0 += 32) {
        v16bf a, bb;
        const float* a0 = &arow[k0 + akb0];        // 8 contiguous floats
        const float* a1 = &arow[k0 + akb1];        // 8 contiguous floats
#pragma unroll
        for (int e = 0; e < 8; ++e) {
            a[e]     = (__bf16)a0[e];
            a[8 + e] = (__bf16)a1[e];
        }
        const float* brow = &W1[u_col * TT + k0 + bkb];  // 16 contiguous floats
#pragma unroll
        for (int e = 0; e < 16; ++e)
            bb[e] = (__bf16)brow[e];
        acc = __builtin_amdgcn_wmma_f32_16x16x32_bf16(
                  false, a, false, bb, (short)0, acc, false, false);
    }

    const float bias = b1[u_col];
#pragma unroll
    for (int r = 0; r < 8; ++r) {
        int mrow = r + (hi ? 8 : 0);
        w1x[(b * 16 + mrow) * TT + u_col] = acc[r] + bias;
    }
}

// ---------------------------------------------------------------------------
// Kernel 3: fused attention.  One block = (b, 16 consecutive t's).
//   w2q tile (16x512) computed in-register via WMMA (never hits HBM),
//   then 192 threads run the 512-long tanh-dot, 16 threads do softmax+output.
// ---------------------------------------------------------------------------
__global__ void attn_kernel(const float* __restrict__ data,
                            const float* __restrict__ query,
                            const float* __restrict__ w1x,
                            const float* __restrict__ W2,
                            const float* __restrict__ b2,
                            const float* __restrict__ Wv,
                            const float* __restrict__ bv,
                            float* __restrict__ out) {
    __shared__ float s_w1x[NN * 520];      // padded rows -> no stride-512 bank conflicts
    __shared__ float s_w2q[16 * 520];
    __shared__ float sWv[TT];
    __shared__ float s_score[16 * NN];

    const int tid = threadIdx.x;
    const int b   = blockIdx.x >> 5;            // 32 t-tiles per batch
    const int t0  = (blockIdx.x & 31) << 4;

    // stage w1x[b] (12x512, already includes b1) and Wv, vectorized float4
    for (int i = tid; i < (NN * TT) / 4; i += blockDim.x) {
        int nrow = i >> 7;                       // / 128 (128 float4 per row)
        int u4   = i & 127;
        *reinterpret_cast<float4*>(&s_w1x[nrow * 520 + u4 * 4]) =
            reinterpret_cast<const float4*>(&w1x[(b * 16 + nrow) * TT])[u4];
    }
    for (int i = tid; i < TT / 4; i += blockDim.x)
        reinterpret_cast<float4*>(sWv)[i] = reinterpret_cast<const float4*>(Wv)[i];

    // --- w2q tile via WMMA: A = query rows t0..t0+15 (K=24, zero-pad to 32) ---
    const int lane = tid & 31;
    const int warp = tid >> 5;
    const bool hi  = lane >= 16;
    const int m    = lane & 15;
    const int akb0 = hi ? 16 : 0;
    const int akb1 = hi ? 24 : 8;
    const int bkb  = hi ? 16 : 0;

    // k in [akb0, akb0+7] is always < 24; only the akb1 range can spill past 24.
    // Clamp index so loads are unconditional; mask the VALUE (v_cndmask, no exec).
    const float* qrow = &query[(b * TT + t0 + m) * QQ];
    v16bf a;
#pragma unroll
    for (int e = 0; e < 8; ++e) {
        int k1       = akb1 + e;
        int k1c      = (k1 < QQ) ? k1 : 0;
        float v0     = qrow[akb0 + e];           // always in range
        float v1     = qrow[k1c];
        a[e]     = (__bf16)v0;
        a[8 + e] = (__bf16)((k1 < QQ) ? v1 : 0.0f);
    }
#pragma unroll
    for (int ut = 0; ut < 4; ++ut) {
        const int u_col = (warp * 4 + ut) * 16 + m;
        const float* w2row = &W2[u_col * QQ];
        v16bf bb;
#pragma unroll
        for (int e = 0; e < 16; ++e) {
            int k   = bkb + e;
            int kc  = (k < QQ) ? k : 0;
            float v = w2row[kc];
            bb[e] = (__bf16)((k < QQ) ? v : 0.0f);
        }
        v8f acc = {};
        acc = __builtin_amdgcn_wmma_f32_16x16x32_bf16(
                  false, a, false, bb, (short)0, acc, false, false);
        const float bias = b2[u_col];
#pragma unroll
        for (int r = 0; r < 8; ++r) {
            int trow = r + (hi ? 8 : 0);
            s_w2q[trow * 520 + u_col] = acc[r] + bias;
        }
    }
    __syncthreads();

    // --- scoring: score[t,n] = sum_u tanh(w1x[n,u] + w2q[t,u]) * Wv[u] + bv ---
    // 4 u per iteration via float4 LDS loads (rows are 16B aligned: 520%4==0).
    if (tid < 16 * NN) {
        const int t_loc = tid / NN, n = tid - t_loc * NN;
        const float4* w1r = reinterpret_cast<const float4*>(&s_w1x[n * 520]);
        const float4* w2r = reinterpret_cast<const float4*>(&s_w2q[t_loc * 520]);
        const float4* wv4 = reinterpret_cast<const float4*>(sWv);
        float acc = 0.0f;
        for (int u4 = 0; u4 < TT / 4; ++u4) {
            float4 p = w1r[u4];
            float4 q = w2r[u4];
            float4 w = wv4[u4];
            acc = __fmaf_rn(tanh_f(p.x + q.x), w.x, acc);
            acc = __fmaf_rn(tanh_f(p.y + q.y), w.y, acc);
            acc = __fmaf_rn(tanh_f(p.z + q.z), w.z, acc);
            acc = __fmaf_rn(tanh_f(p.w + q.w), w.w, acc);
        }
        s_score[tid] = acc + bv[0];
    }
    __syncthreads();

    // --- softmax over n (12) + scale input ---
    if (tid < 16) {
        const int t = t0 + tid;
        float sc[NN];
        float mx = -1e30f;
#pragma unroll
        for (int n = 0; n < NN; ++n) { sc[n] = s_score[tid * NN + n]; mx = fmaxf(mx, sc[n]); }
        float sum = 0.0f;
#pragma unroll
        for (int n = 0; n < NN; ++n) { sc[n] = __expf(sc[n] - mx); sum += sc[n]; }
        const float inv = fast_rcp(sum);
#pragma unroll
        for (int n = 0; n < NN; ++n)
            out[(b * TT + t) * NN + n] = data[(b * TT + t) * NN + n] * sc[n] * inv;
    }
}

// ---------------------------------------------------------------------------
extern "C" void kernel_launch(void* const* d_in, const int* in_sizes, int n_in,
                              void* d_out, int out_size, void* d_ws, size_t ws_size,
                              hipStream_t stream) {
    (void)in_sizes; (void)n_in; (void)out_size; (void)ws_size;
    const float* data  = (const float*)d_in[0];
    // d_in[1]=h0, d_in[2]=c0 (always zero), d_in[3]=n : unused
    const float* W_ih1 = (const float*)d_in[4];
    const float* b_ih1 = (const float*)d_in[6];
    const float* b_hh1 = (const float*)d_in[7];
    const float* W_ih2 = (const float*)d_in[8];
    const float* b_ih2 = (const float*)d_in[10];
    const float* b_hh2 = (const float*)d_in[11];
    const float* W1    = (const float*)d_in[12];
    const float* b1    = (const float*)d_in[13];
    const float* W2    = (const float*)d_in[14];
    const float* b2    = (const float*)d_in[15];
    const float* Wv    = (const float*)d_in[16];
    const float* bv    = (const float*)d_in[17];
    float* out = (float*)d_out;

    // workspace: query (B,T,24) then w1x (B,16,T)  -> ~10.5 MB
    float* ws_query = (float*)d_ws;
    float* ws_w1x   = ws_query + (size_t)BB * TT * QQ;

    lstm_query_kernel<<<(BB * TT) / 256, 256, 0, stream>>>(
        data, W_ih1, b_ih1, b_hh1, W_ih2, b_ih2, b_hh2, ws_query);
    w1x_kernel<<<BB * 8, 128, 0, stream>>>(data, W1, b1, ws_w1x);
    attn_kernel<<<BB * 32, 256, 0, stream>>>(
        data, ws_query, ws_w1x, W2, b2, Wv, bv, out);
}
